// GraphSAGE_13477607375472
// MI455X (gfx1250) — matrix operand order repaired and checked by reading, not compiled, not run on hardware
//
#include <hip/hip_runtime.h>
#include <hip/hip_bf16.h>

#define N_NODES 50000
#define N_EDGES 800000
#define HID 128
#define NLAYER 3
#define HCAT (HID * NLAYER)   // 384
#define FC_HID 256
#define N_CLASS 10
#define N_GRAPHS 64
#define NTILES (N_NODES / 16) // 3125 exactly

typedef __attribute__((ext_vector_type(16))) __bf16 v16bf;
typedef __attribute__((ext_vector_type(8)))  float  v8f;

// ---------------------------------------------------------------- utilities

__global__ void set_int_kernel(int* p, int v) {
    if (threadIdx.x == 0 && blockIdx.x == 0) *p = v;
}

__global__ void zero_f32(float* __restrict__ p, int n) {
    int i = blockIdx.x * blockDim.x + threadIdx.x;
    if (i < n) p[i] = 0.0f;
}

// min over edge_index (reference subtracts the global min)
__global__ void min_reduce(const int* __restrict__ v, int n, int* __restrict__ out) {
    __shared__ int sm;
    if (threadIdx.x == 0) sm = 0x7fffffff;
    __syncthreads();
    int m = 0x7fffffff;
    for (int i = blockIdx.x * blockDim.x + threadIdx.x; i < n;
         i += gridDim.x * blockDim.x)
        m = min(m, v[i]);
    atomicMin(&sm, m);
    __syncthreads();
    if (threadIdx.x == 0) atomicMin(out, sm);
}

__global__ void deg_kernel(const int* __restrict__ dst, const int* __restrict__ minv,
                           float* __restrict__ deg) {
    int e = blockIdx.x * blockDim.x + threadIdx.x;
    if (e < N_EDGES) atomicAdd(&deg[dst[e] - *minv], 1.0f);
}

__global__ void finalize_deg(float* __restrict__ deg) {
    int i = blockIdx.x * blockDim.x + threadIdx.x;
    if (i < N_NODES) deg[i] = 1.0f / fmaxf(deg[i], 1.0f);
}

__global__ void count_kernel(const int* __restrict__ batch, float* __restrict__ counts) {
    int i = blockIdx.x * blockDim.x + threadIdx.x;
    if (i < N_NODES) atomicAdd(&counts[batch[i]], 1.0f);
}

// -------------------------------------------------- weight prep (B fragments)
// Pack Wl/Wr into WMMA B-operand fragment order, bf16, lane-major:
// fragment f = ((layer*2+mat)*8 + ctile)*4 + kstep ; element [lane*16 + e]
// B layout (32x16 bf16): lane L holds col N = L%16; K = 32*kstep + 8*(L/16) + (e%8) + 16*(e/8)
__global__ void prep_w(const float* __restrict__ Wl, const float* __restrict__ Wr,
                       __bf16* __restrict__ frag) {
    int tid = blockIdx.x * blockDim.x + threadIdx.x;
    if (tid >= NLAYER * 2 * 8 * 4 * 32 * 16) return;
    int e = tid & 15;
    int L = (tid >> 4) & 31;
    int s = (tid >> 9) & 3;
    int c = (tid >> 11) & 7;
    int m = (tid >> 14) & 1;
    int l = tid >> 15;
    int K   = 32 * s + 8 * (L >> 4) + (e & 7) + 16 * (e >> 3);
    int col = c * 16 + (L & 15);
    const float* W = (m == 0 ? Wl : Wr) + (size_t)l * HID * HID;
    int f = ((l * 2 + m) * 8 + c) * 4 + s;
    frag[(size_t)f * 512 + L * 16 + e] = (__bf16)W[K * HID + col];
}

// ------------------------------------------------------------ edge scatter
// agg[dst] += h[src]; one thread = one edge x 4 feature floats
__global__ void scatter_kernel(const float* __restrict__ h, int hstride,
                               const int* __restrict__ src, const int* __restrict__ dst,
                               const int* __restrict__ minv, float* __restrict__ agg) {
    unsigned id = blockIdx.x * blockDim.x + threadIdx.x;
    if (id >= (unsigned)N_EDGES * 32u) return;
    int e = id >> 5;
    int c = (id & 31) * 4;
    int mv = *minv;
    int sn = src[e] - mv;
    int dn = dst[e] - mv;
    const float4 v = *reinterpret_cast<const float4*>(h + (size_t)sn * hstride + c);
    float* ap = agg + (size_t)dn * HID + c;
    atomicAdd(ap + 0, v.x);
    atomicAdd(ap + 1, v.y);
    atomicAdd(ap + 2, v.z);
    atomicAdd(ap + 3, v.w);
}

// ------------------------------------------------------------- WMMA GEMM
__device__ __forceinline__ float4 scale4(float4 v, float s) {
    return make_float4(v.x * s, v.y * s, v.z * s, v.w * s);
}
__device__ __forceinline__ v16bf pack_bf16(float4 a, float4 b, float4 c, float4 d) {
    v16bf r;
    r[0]  = (__bf16)a.x; r[1]  = (__bf16)a.y; r[2]  = (__bf16)a.z; r[3]  = (__bf16)a.w;
    r[4]  = (__bf16)b.x; r[5]  = (__bf16)b.y; r[6]  = (__bf16)b.z; r[7]  = (__bf16)b.w;
    r[8]  = (__bf16)c.x; r[9]  = (__bf16)c.y; r[10] = (__bf16)c.z; r[11] = (__bf16)c.w;
    r[12] = (__bf16)d.x; r[13] = (__bf16)d.y; r[14] = (__bf16)d.z; r[15] = (__bf16)d.w;
    return r;
}

// h_out = relu((agg*inv_deg) @ Wl + h @ Wr + bl), written into hcat column band.
// One wave = 16 node rows x 128 output cols (8 C-tiles), K = 128 in 4 steps of 32.
__global__ void __launch_bounds__(256)
sage_gemm(const float* __restrict__ h, int hstride,
          const float* __restrict__ agg,
          const float* __restrict__ invdeg,
          const __bf16* __restrict__ bfrags, int layer,
          const float* __restrict__ bias,
          float* __restrict__ outc) {   // row stride HCAT
    int wave = threadIdx.x >> 5;
    int lane = threadIdx.x & 31;
    int tile = blockIdx.x * 8 + wave;
    if (tile >= NTILES) return;          // wave-uniform: EXEC stays all-1s

    int row  = lane & 15;
    int half = lane >> 4;
    int node = tile * 16 + row;
    float sc = invdeg[node];
    const float* hrow = h   + (size_t)node * hstride;
    const float* arow = agg + (size_t)node * HID;

    v8f acc[8];
#pragma unroll
    for (int c = 0; c < 8; ++c) acc[c] = (v8f){0, 0, 0, 0, 0, 0, 0, 0};

    const __bf16* bl_base = bfrags + (size_t)((layer * 2 + 0) * 8) * 4 * 512;
    const __bf16* br_base = bfrags + (size_t)((layer * 2 + 1) * 8) * 4 * 512;

#pragma unroll
    for (int s = 0; s < 4; ++s) {
        // A layout (16x32 bf16): lane half 0 holds K [k0..k0+7] and [k0+16..k0+23]
        int k0 = 32 * s + 8 * half;
        float4 h0 = *(const float4*)(hrow + k0);
        float4 h1 = *(const float4*)(hrow + k0 + 4);
        float4 h2 = *(const float4*)(hrow + k0 + 16);
        float4 h3 = *(const float4*)(hrow + k0 + 20);
        float4 a0 = scale4(*(const float4*)(arow + k0),      sc);
        float4 a1 = scale4(*(const float4*)(arow + k0 + 4),  sc);
        float4 a2 = scale4(*(const float4*)(arow + k0 + 16), sc);
        float4 a3 = scale4(*(const float4*)(arow + k0 + 20), sc);
        v16bf Ah = pack_bf16(h0, h1, h2, h3);
        v16bf Aa = pack_bf16(a0, a1, a2, a3);

#pragma unroll
        for (int c = 0; c < 8; ++c) {
            v16bf Bl = *(const v16bf*)(bl_base + ((size_t)(c * 4 + s)) * 512 + lane * 16);
            acc[c] = __builtin_amdgcn_wmma_f32_16x16x32_bf16(
                false, Aa, false, Bl, (short)0, acc[c], false, false);
            v16bf Br = *(const v16bf*)(br_base + ((size_t)(c * 4 + s)) * 512 + lane * 16);
            acc[c] = __builtin_amdgcn_wmma_f32_16x16x32_bf16(
                false, Ah, false, Br, (short)0, acc[c], false, false);
        }
    }

    // C/D layout: VGPR r -> M = 8*half + r ; N = lane%16
    int ncol = lane & 15;
#pragma unroll
    for (int c = 0; c < 8; ++c) {
        int col = c * 16 + ncol;
        float b = bias[col];
#pragma unroll
        for (int r = 0; r < 8; ++r) {
            float v = acc[c][r] + b;
            v = v > 0.0f ? v : 0.0f;
            int nn = tile * 16 + 8 * half + r;
            outc[(size_t)nn * HCAT + col] = v;
        }
    }
}

// ------------------------------------------------------------ pooling + MLP
__global__ void pool_kernel(const float* __restrict__ hcat, const int* __restrict__ batch,
                            float* __restrict__ pooled) {
    unsigned id = blockIdx.x * blockDim.x + threadIdx.x;
    if (id >= (unsigned)N_NODES * (HCAT / 4)) return;
    int n = id / (HCAT / 4);
    int c = (id % (HCAT / 4)) * 4;
    int g = batch[n];
    const float4 v = *reinterpret_cast<const float4*>(hcat + (size_t)n * HCAT + c);
    float* p = pooled + (size_t)g * HCAT + c;
    atomicAdd(p + 0, v.x);
    atomicAdd(p + 1, v.y);
    atomicAdd(p + 2, v.z);
    atomicAdd(p + 3, v.w);
}

__global__ void __launch_bounds__(256)
fc_kernel(const float* __restrict__ pooled, const float* __restrict__ counts,
          const float* __restrict__ fc1w, const float* __restrict__ fc1b,
          const float* __restrict__ fc2w, const float* __restrict__ fc2b,
          float* __restrict__ out) {
    __shared__ float sp[HCAT];
    __shared__ float s1[FC_HID];
    int g = blockIdx.x;
    float invc = 1.0f / fmaxf(counts[g], 1.0f);
    for (int i = threadIdx.x; i < HCAT; i += blockDim.x)
        sp[i] = pooled[(size_t)g * HCAT + i] * invc;
    __syncthreads();
    for (int j = threadIdx.x; j < FC_HID; j += blockDim.x) {
        float a = fc1b[j];
        for (int k = 0; k < HCAT; ++k) a += sp[k] * fc1w[(size_t)k * FC_HID + j];
        s1[j] = fmaxf(a, 0.0f);
    }
    __syncthreads();
    if (threadIdx.x < N_CLASS) {
        int j = threadIdx.x;
        float a = fc2b[j];
        for (int k = 0; k < FC_HID; ++k) a += s1[k] * fc2w[(size_t)k * N_CLASS + j];
        out[(size_t)g * N_CLASS + j] = a;
    }
}

// ------------------------------------------------------------------ driver

extern "C" void kernel_launch(void* const* d_in, const int* in_sizes, int n_in,
                              void* d_out, int out_size, void* d_ws, size_t ws_size,
                              hipStream_t stream) {
    (void)in_sizes; (void)n_in; (void)out_size; (void)ws_size;
    const float* x     = (const float*)d_in[0];
    const int*   ei    = (const int*)d_in[1];
    const int*   batch = (const int*)d_in[2];
    const float* Wl    = (const float*)d_in[3];
    const float* Wr    = (const float*)d_in[4];
    const float* bl    = (const float*)d_in[5];
    const float* fc1w  = (const float*)d_in[6];
    const float* fc1b  = (const float*)d_in[7];
    const float* fc2w  = (const float*)d_in[8];
    const float* fc2b  = (const float*)d_in[9];
    float* out = (float*)d_out;

    char* ws = (char*)d_ws;
    size_t off = 0;
    auto take = [&](size_t bytes) -> void* {
        void* p = ws + off;
        off += (bytes + 255) & ~(size_t)255;
        return p;
    };
    float*  hcat   = (float*)take((size_t)N_NODES * HCAT * sizeof(float));
    float*  agg    = (float*)take((size_t)N_NODES * HID * sizeof(float));
    float*  invdeg = (float*)take((size_t)N_NODES * sizeof(float));
    float*  pooled = (float*)take((size_t)N_GRAPHS * HCAT * sizeof(float));
    float*  counts = (float*)take((size_t)N_GRAPHS * sizeof(float));
    __bf16* bfrags = (__bf16*)take((size_t)NLAYER * 2 * 8 * 4 * 512 * sizeof(__bf16));
    int*    minv   = (int*)take(sizeof(int));

    set_int_kernel<<<1, 1, 0, stream>>>(minv, 0x7fffffff);
    min_reduce<<<512, 256, 0, stream>>>(ei, 2 * N_EDGES, minv);

    zero_f32<<<(N_NODES + 255) / 256, 256, 0, stream>>>(invdeg, N_NODES);
    zero_f32<<<(N_GRAPHS * HCAT + 255) / 256, 256, 0, stream>>>(pooled, N_GRAPHS * HCAT);
    zero_f32<<<1, 64, 0, stream>>>(counts, N_GRAPHS);

    deg_kernel<<<(N_EDGES + 255) / 256, 256, 0, stream>>>(ei + N_EDGES, minv, invdeg);
    finalize_deg<<<(N_NODES + 255) / 256, 256, 0, stream>>>(invdeg);
    count_kernel<<<(N_NODES + 255) / 256, 256, 0, stream>>>(batch, counts);

    prep_w<<<(NLAYER * 2 * 8 * 4 * 32 * 16) / 256, 256, 0, stream>>>(Wl, Wr, bfrags);

    for (int l = 0; l < NLAYER; ++l) {
        const float* h = (l == 0) ? x : (hcat + (size_t)(l - 1) * HID);
        int hs = (l == 0) ? HID : HCAT;
        zero_f32<<<(N_NODES * HID + 255) / 256, 256, 0, stream>>>(agg, N_NODES * HID);
        scatter_kernel<<<(N_EDGES * 32) / 256, 256, 0, stream>>>(h, hs, ei, ei + N_EDGES,
                                                                 minv, agg);
        sage_gemm<<<(NTILES + 7) / 8, 256, 0, stream>>>(h, hs, agg, invdeg, bfrags, l,
                                                        bl + (size_t)l * HID,
                                                        hcat + (size_t)l * HID);
    }

    pool_kernel<<<((unsigned)N_NODES * (HCAT / 4) + 255) / 256, 256, 0, stream>>>(
        hcat, batch, pooled);
    fc_kernel<<<N_GRAPHS, 256, 0, stream>>>(pooled, counts, fc1w, fc1b, fc2w, fc2b, out);
}
